// HierarchicalCodeEncoder_1468878815517
// MI455X (gfx1250) — compile-verified, matrix operand order checked by reading, not run on hardware
//
#include <hip/hip_runtime.h>

// ---------------------------------------------------------------------------
// Types for CDNA5 WMMA (wave32, 16x16x32 bf16 -> f32 accum)
// ---------------------------------------------------------------------------
typedef __bf16        v16bf __attribute__((ext_vector_type(16)));
typedef float         v8f   __attribute__((ext_vector_type(8)));
typedef unsigned int  v8u   __attribute__((ext_vector_type(8)));

#define BM 128
#define BN 128
#define BK 32
#define LDSTR 34   // ushort stride per LDS row (BK + 2 pad, keeps rows 4B aligned)

__device__ __forceinline__ unsigned short f32_to_bf16(float f) {
  union { float f; unsigned int u; } x; x.f = f;
  unsigned int r = x.u + 0x7FFFu + ((x.u >> 16) & 1u);   // round-to-nearest-even
  return (unsigned short)(r >> 16);
}

union FragU { v8u u; v16bf bf; };

// Load a 16x32 bf16 fragment from LDS per the CDNA5 16-bit A-matrix layout:
//   lanes 0-15 (row M=lane):  v0..v3 = K 0..7 (pairs), v4..v7 = K 16..23
//   lanes 16-31 (row M=lane-16): v0..v3 = K 8..15,     v4..v7 = K 24..31
__device__ __forceinline__ v16bf load_frag16(const unsigned short* tile, int lane) {
  const int half = lane >> 4;
  const int m    = lane & 15;
  const unsigned int* row = (const unsigned int*)(tile + m * LDSTR);
  FragU f;
#pragma unroll
  for (int i = 0; i < 8; ++i) {
    const int k = (i < 4 ? 0 : 16) + half * 8 + (i & 3) * 2;
    f.u[i] = row[k >> 1];
  }
  return f.bf;
}

// ---------------------------------------------------------------------------
// token_emb = hidden @ W + b   (M=B*S=65536, N=K=768), bf16 WMMA, f32 accum
// ---------------------------------------------------------------------------
__global__ __launch_bounds__(256)
void gemm_bias_bf16wmma(const float* __restrict__ A,   // [M][K] fp32
                        const float* __restrict__ W,   // [K][N] fp32
                        const float* __restrict__ bias,// [N]
                        float* __restrict__ C,         // [M][N] fp32
                        int M, int N, int K) {
  __shared__ unsigned short lA[BM * LDSTR];  // A tile, row-major [m][k], bf16
  __shared__ unsigned short lB[BN * LDSTR];  // W tile transposed: [n][k], bf16

  const int tid  = threadIdx.x;
  const int lane = tid & 31;
  const int wid  = tid >> 5;     // 8 waves
  const int wm   = wid & 3;      // 4 waves along M (32 rows each)
  const int wn   = wid >> 2;     // 2 waves along N (64 cols each)

  const int n0 = blockIdx.x * BN;
  const int m0 = blockIdx.y * BM;

  v8f acc[2][4] = {};            // 2 M-tiles x 4 N-tiles of 16x16

  for (int k0 = 0; k0 < K; k0 += BK) {
    // ---- stage A tile (BM x BK), fp32 -> bf16 -----------------------------
    {
      const int row = tid >> 3;        // 0..31
      const int kc  = (tid & 7) * 4;   // 0,4,...,28
#pragma unroll
      for (int r = 0; r < 4; ++r) {
        const int rr = row + r * 32;
        const float4 v = *(const float4*)(&A[(size_t)(m0 + rr) * K + k0 + kc]);
        unsigned short* dst = &lA[rr * LDSTR + kc];
        dst[0] = f32_to_bf16(v.x);
        dst[1] = f32_to_bf16(v.y);
        dst[2] = f32_to_bf16(v.z);
        dst[3] = f32_to_bf16(v.w);
      }
    }
    // ---- stage W tile (BK x BN) transposed into [n][k] --------------------
    {
      const int krow = tid >> 5;       // 0..7
      const int nc   = (tid & 31) * 4; // 0..124
#pragma unroll
      for (int r = 0; r < 4; ++r) {
        const int kk = krow + r * 8;
        const float4 v = *(const float4*)(&W[(size_t)(k0 + kk) * N + n0 + nc]);
        lB[(nc + 0) * LDSTR + kk] = f32_to_bf16(v.x);
        lB[(nc + 1) * LDSTR + kk] = f32_to_bf16(v.y);
        lB[(nc + 2) * LDSTR + kk] = f32_to_bf16(v.z);
        lB[(nc + 3) * LDSTR + kk] = f32_to_bf16(v.w);
      }
    }
    __syncthreads();

    v16bf afr[2], bfr[4];
#pragma unroll
    for (int a = 0; a < 2; ++a)
      afr[a] = load_frag16(&lA[(wm * 32 + a * 16) * LDSTR], lane);
#pragma unroll
    for (int b = 0; b < 4; ++b)
      bfr[b] = load_frag16(&lB[(wn * 64 + b * 16) * LDSTR], lane);

#pragma unroll
    for (int a = 0; a < 2; ++a)
#pragma unroll
      for (int b = 0; b < 4; ++b)
        acc[a][b] = __builtin_amdgcn_wmma_f32_16x16x32_bf16(
            /*neg_a=*/false, afr[a], /*neg_b=*/false, bfr[b],
            /*c_mod=*/(short)0, acc[a][b], /*reuse_a=*/false, /*reuse_b=*/false);

    __syncthreads();
  }

  // ---- epilogue: bias add, store fp32 -------------------------------------
  // C/D layout: VGPR j -> M = j (lanes 0-15) or j+8 (lanes 16-31); N = lane&15
  const int half = lane >> 4;
  const int nl   = lane & 15;
#pragma unroll
  for (int a = 0; a < 2; ++a) {
#pragma unroll
    for (int b = 0; b < 4; ++b) {
      const int gn = n0 + wn * 64 + b * 16 + nl;
      const float bv = bias[gn];
#pragma unroll
      for (int j = 0; j < 8; ++j) {
        const int gm = m0 + wm * 32 + a * 16 + j + 8 * half;
        C[(size_t)gm * N + gn] = acc[a][b][j] + bv;
      }
    }
  }
}

// ---------------------------------------------------------------------------
// Segment mean: out[b, i, :] = mean(x[b, start:end, :]) or 0 when start>=end.
// One block per (b, i); D=768 -> 3 elements per thread, coalesced over D.
// ---------------------------------------------------------------------------
__global__ __launch_bounds__(256)
void segment_mean_kernel(const float* __restrict__ x,      // [B][S][768]
                         const int*   __restrict__ bounds, // [B][N][2]
                         float*       __restrict__ out,    // [B][N][768]
                         int S, int N) {
  const int b = blockIdx.x / N;
  const int i = blockIdx.x - b * N;
  const int start = bounds[((size_t)b * N + i) * 2 + 0];
  const int end   = bounds[((size_t)b * N + i) * 2 + 1];
  const int tid = threadIdx.x;

  float a0 = 0.f, a1 = 0.f, a2 = 0.f;
  const float* base = x + (size_t)b * S * 768;
  for (int s = start; s < end; ++s) {
    const float* row = base + (size_t)s * 768;
    a0 += row[tid];
    a1 += row[tid + 256];
    a2 += row[tid + 512];
  }
  const int len = end - start;
  const float scale = (len > 0) ? 1.0f / (float)len : 0.0f;
  float* o = out + ((size_t)b * N + i) * 768;
  o[tid]       = a0 * scale;
  o[tid + 256] = a1 * scale;
  o[tid + 512] = a2 * scale;
}

// ---------------------------------------------------------------------------
// file_emb[b, d] = mean over 64 functions of func_emb[b, f, d]
// ---------------------------------------------------------------------------
__global__ __launch_bounds__(256)
void file_mean_kernel(const float* __restrict__ func,  // [B][64][768]
                      float*       __restrict__ out) { // [B][768]
  const int idx = blockIdx.x * 256 + threadIdx.x;      // over B*768 = 24576
  const int b = idx / 768;
  const int d = idx - b * 768;
  float s = 0.f;
#pragma unroll 8
  for (int f = 0; f < 64; ++f)
    s += func[((size_t)b * 64 + f) * 768 + d];
  out[idx] = s * (1.0f / 64.0f);
}

// ---------------------------------------------------------------------------
extern "C" void kernel_launch(void* const* d_in, const int* in_sizes, int n_in,
                              void* d_out, int out_size, void* d_ws, size_t ws_size,
                              hipStream_t stream) {
  (void)in_sizes; (void)n_in; (void)out_size; (void)d_ws; (void)ws_size;

  const float* hidden = (const float*)d_in[0];   // [32][2048][768] fp32
  const float* projW  = (const float*)d_in[1];   // [768][768] fp32
  const float* projB  = (const float*)d_in[2];   // [768] fp32
  const int*   lineB  = (const int*)d_in[3];     // [32][512][2] int32
  const int*   funcB  = (const int*)d_in[4];     // [32][64][2] int32

  const int B = 32, S = 2048, D = 768, NL = 512, NF = 64;

  float* out   = (float*)d_out;
  float* token = out;                               // [B][S][D]
  float* line  = token + (size_t)B * S  * D;        // [B][NL][D]
  float* func  = line  + (size_t)B * NL * D;        // [B][NF][D]
  float* file  = func  + (size_t)B * NF * D;        // [B][D]

  // 1) token_emb = hidden @ W + b  (WMMA bf16, fp32 accum)
  dim3 ggrid(D / BN, (B * S) / BM);                 // (6, 512)
  gemm_bias_bf16wmma<<<ggrid, 256, 0, stream>>>(hidden, projW, projB, token,
                                                B * S, D, D);

  // 2) line_emb = segment_mean(token_emb, line_boundaries)
  segment_mean_kernel<<<B * NL, 256, 0, stream>>>(token, lineB, line, S, NL);

  // 3) func_emb = segment_mean(line_emb, function_boundaries)
  segment_mean_kernel<<<B * NF, 256, 0, stream>>>(line, funcB, func, NL, NF);

  // 4) file_emb = mean over function axis
  file_mean_kernel<<<(B * D) / 256, 256, 0, stream>>>(func, file);
}
